// AdaptPoint_Augmentor_55954833933048
// MI455X (gfx1250) — matrix operand order repaired and checked by reading, not compiled, not run on hardware
//
#include <hip/hip_runtime.h>
#include <hip/hip_bf16.h>

#define BB    64
#define NAA   4
#define NPP   8192
#define CC    256
#define KK    24
#define HEADS 4
#define HDD   64
#define ROWS  256           // B * NA
#define EPSB  1e-5f

typedef __attribute__((ext_vector_type(2))) float v2f;
typedef __attribute__((ext_vector_type(8))) float v8f;

#define HAVE_WMMA_F32 __has_builtin(__builtin_amdgcn_wmma_f32_16x16x4_f32)
#if !HAVE_WMMA_F32
typedef __attribute__((ext_vector_type(16))) __bf16 v16bf;
#endif

// ---------------------------------------------------------------------------
// Kernel 1: per-(batch,anchor) KNN (top-24 smallest sq-dist) + gather + maxpool
// grid = B*NA blocks, 256 threads. Distances staged in LDS (32KB of 320KB/WGP).
// ---------------------------------------------------------------------------
__global__ __launch_bounds__(256) void knn_gather_max_kernel(
    const float* __restrict__ a_points,   // [B, NA, 3]
    const float* __restrict__ sa_xyz,     // [B, NP, 3]
    const float* __restrict__ sa_x,       // [B, NP, C]
    float* __restrict__ local_feat)       // [ROWS, C]
{
    __shared__ float dist[NPP];
    __shared__ float rmin[256];
    __shared__ int   rarg[256];
    __shared__ int   kidx[KK];

    const int tid = threadIdx.x;
    const int b = blockIdx.x / NAA;
    const int m = blockIdx.x % NAA;

    const float ax = a_points[(b * NAA + m) * 3 + 0];
    const float ay = a_points[(b * NAA + m) * 3 + 1];
    const float az = a_points[(b * NAA + m) * 3 + 2];

    for (int p = tid; p < NPP; p += 256) {
        const float dx = ax - sa_xyz[(b * NPP + p) * 3 + 0];
        const float dy = ay - sa_xyz[(b * NPP + p) * 3 + 1];
        const float dz = az - sa_xyz[(b * NPP + p) * 3 + 2];
        dist[p] = dx * dx + dy * dy + dz * dz;
    }
    __syncthreads();

    for (int kk = 0; kk < KK; ++kk) {
        float best = 3.4e38f; int bi = NPP;
        for (int p = tid; p < NPP; p += 256) {
            const float v = dist[p];
            if (v < best) { best = v; bi = p; }
        }
        rmin[tid] = best; rarg[tid] = bi;
        __syncthreads();
        for (int s = 128; s > 0; s >>= 1) {
            if (tid < s) {
                const float ov = rmin[tid + s]; const int oi = rarg[tid + s];
                if (ov < rmin[tid] || (ov == rmin[tid] && oi < rarg[tid])) {
                    rmin[tid] = ov; rarg[tid] = oi;
                }
            }
            __syncthreads();
        }
        if (tid == 0) { kidx[kk] = rarg[0]; dist[rarg[0]] = 3.4e38f; }
        __syncthreads();
    }

    // gather + max over K neighbors; thread = channel
    float mx = -3.4e38f;
    for (int kk = 0; kk < KK; ++kk) {
        const int p = kidx[kk];
        if (kk + 1 < KK)
            __builtin_prefetch(&sa_x[(size_t)(b * NPP + kidx[kk + 1]) * CC + tid], 0, 1);
        mx = fmaxf(mx, sa_x[(size_t)(b * NPP + p) * CC + tid]);
    }
    local_feat[blockIdx.x * CC + tid] = mx;
}

// ---------------------------------------------------------------------------
// Kernel 2: rxyz_pre[r,c] = (a-mean)·pos_w[c] + pos_b[c] ; glob_pre = a·glob_w[c]
// ---------------------------------------------------------------------------
__global__ __launch_bounds__(256) void pre_kernel(
    const float* __restrict__ a_points,
    const float* __restrict__ pos_w, const float* __restrict__ pos_b,
    const float* __restrict__ glob_w,
    float* __restrict__ rxyz_pre, float* __restrict__ glob_pre)
{
    const int r = blockIdx.x, c = threadIdx.x;
    const int b = r / NAA, m = r % NAA;
    float mean0 = 0.f, mean1 = 0.f, mean2 = 0.f;
    for (int j = 0; j < NAA; ++j) {
        mean0 += a_points[(b * NAA + j) * 3 + 0];
        mean1 += a_points[(b * NAA + j) * 3 + 1];
        mean2 += a_points[(b * NAA + j) * 3 + 2];
    }
    mean0 *= 0.25f; mean1 *= 0.25f; mean2 *= 0.25f;
    const float ax = a_points[(b * NAA + m) * 3 + 0];
    const float ay = a_points[(b * NAA + m) * 3 + 1];
    const float az = a_points[(b * NAA + m) * 3 + 2];
    const float rx = ax - mean0, ry = ay - mean1, rz = az - mean2;
    rxyz_pre[r * CC + c] = rx * pos_w[c * 3 + 0] + ry * pos_w[c * 3 + 1]
                         + rz * pos_w[c * 3 + 2] + pos_b[c];
    glob_pre[r * CC + c] = ax * glob_w[c * 3 + 0] + ay * glob_w[c * 3 + 1]
                         + az * glob_w[c * 3 + 2];
}

// ---------------------------------------------------------------------------
// BN statistics over ROWS rows -> per-channel scale/shift (population var)
// ---------------------------------------------------------------------------
__global__ __launch_bounds__(256) void bnstats_kernel(
    const float* __restrict__ x, int nch,
    const float* __restrict__ gamma, const float* __restrict__ beta,
    float* __restrict__ scale, float* __restrict__ shift)
{
    const int c = blockIdx.x * blockDim.x + threadIdx.x;
    if (c >= nch) return;
    float s = 0.f, sq = 0.f;
    for (int r = 0; r < ROWS; ++r) {
        const float v = x[r * nch + c];
        s += v; sq += v * v;
    }
    const float mean = s * (1.0f / ROWS);
    const float var = sq * (1.0f / ROWS) - mean * mean;
    const float sc = gamma[c] * rsqrtf(var + EPSB);
    scale[c] = sc;
    shift[c] = beta[c] - mean * sc;
}

// ---------------------------------------------------------------------------
// WMMA GEMM: C[M,N] = A[M,K] * op(B) (+ bias per column). One wave per 16x16
// tile. K unrolled by 16 (4 x V_WMMA_F32_16X16X4_F32 per iteration) so all 8
// fragment loads issue before the WMMA batch (one loadcnt wait per 4 MMAs).
// transB / bias are compile-time to keep the inner loop branch-free.
// ---------------------------------------------------------------------------
template <int TRANSB, int HAS_BIAS>
__global__ __launch_bounds__(32) void wmma_gemm_kernel(
    const float* __restrict__ A, const float* __restrict__ Bm,
    const float* __restrict__ bias, float* __restrict__ Cm,
    int N, int Kd, int lda, int ldb, int ldc)
{
    const int tilesN = N >> 4;
    const int tM = blockIdx.x / tilesN;
    const int tN = blockIdx.x % tilesN;
    const int lane = threadIdx.x;
    const int half = lane >> 4;      // 0: lanes 0-15, 1: lanes 16-31
    const int l    = lane & 15;
    const int ar = tM * 16 + l;      // A row for this lane
    const int bc = tN * 16 + l;      // B/C column for this lane

    const float* __restrict__ Arow = A + (size_t)ar * lda;
    const float* __restrict__ Bcol = TRANSB ? (Bm + (size_t)bc * ldb) : (Bm + bc);

    v8f acc = {};
#if HAVE_WMMA_F32
    for (int k0 = 0; k0 < Kd; k0 += 16) {
        v2f a[4], bfrag[4];
#pragma unroll
        for (int u = 0; u < 4; ++u) {
            const int k = k0 + 4 * u + half * 2;
            // A 16x4 fragment: V0 = K(2*half), V1 = K(2*half+1); 8B-aligned pair
            a[u] = *(const v2f*)(Arow + k);
            if (TRANSB) {
                bfrag[u] = *(const v2f*)(Bcol + k);
            } else {
                bfrag[u].x = Bcol[(size_t)(k + 0) * ldb];
                bfrag[u].y = Bcol[(size_t)(k + 1) * ldb];
            }
        }
#pragma unroll
        for (int u = 0; u < 4; ++u)
            acc = __builtin_amdgcn_wmma_f32_16x16x4_f32(
                false, a[u], false, bfrag[u], (short)0, acc, false, false);
    }
#else
    for (int k0 = 0; k0 < Kd; k0 += 32) {
        v16bf af, bfr;
        for (int e = 0; e < 16; ++e) {
            const int ka = k0 + e + 8 * half + (e >= 8 ? 8 : 0);
            af[e] = (__bf16)Arow[ka];
            const int kb = k0 + e + 8 * half + (e >= 8 ? 8 : 0);
            const float bv = TRANSB ? Bcol[kb] : Bcol[(size_t)kb * ldb];
            bfr[e] = (__bf16)bv;
        }
        acc = __builtin_amdgcn_wmma_f32_16x16x32_bf16(
            false, af, false, bfr, (short)0, acc, false, false);
    }
#endif
    const float bv = HAS_BIAS ? bias[bc] : 0.0f;
#pragma unroll
    for (int vi = 0; vi < 8; ++vi) {
        const int row = tM * 16 + vi + half * 8;
        Cm[(size_t)row * ldc + bc] = acc[vi] + bv;
    }
}

// ---------------------------------------------------------------------------
// Attention over NA=4 tokens, per batch. thread = channel (h = tid/64).
// ---------------------------------------------------------------------------
__global__ __launch_bounds__(256) void attn_kernel(
    const float* __restrict__ qkv,        // [ROWS, 3C]
    const float* __restrict__ rxyz_pre,   // [ROWS, C]
    const float* __restrict__ pos_sc, const float* __restrict__ pos_sh,
    float* __restrict__ o_attn)           // [ROWS, C]
{
    __shared__ float red[256];
    __shared__ float logit[HEADS][NAA][NAA];
    const int tid = threadIdx.x;
    const int b = blockIdx.x;
    const int h = tid >> 6, d = tid & 63;

    float q[NAA], k[NAA], v[NAA];
    for (int m = 0; m < NAA; ++m) {
        const int r = b * NAA + m;
        const float rxn = rxyz_pre[r * CC + tid] * pos_sc[tid] + pos_sh[tid];
        q[m] = qkv[r * (3 * CC) + 0 * CC + tid] + rxn;
        k[m] = qkv[r * (3 * CC) + 1 * CC + tid] + rxn;
        v[m] = qkv[r * (3 * CC) + 2 * CC + tid] + rxn;
    }
    for (int m = 0; m < NAA; ++m)
        for (int n = 0; n < NAA; ++n) {
            red[tid] = q[m] * k[n];
            __syncthreads();
            for (int s = 32; s > 0; s >>= 1) {
                if (d < s) red[tid] += red[tid + s];
                __syncthreads();
            }
            if (d == 0) logit[h][m][n] = red[tid];
            __syncthreads();
        }
    const float sc = 0.125f;  // 1/sqrt(HD=64)
    for (int m = 0; m < NAA; ++m) {
        float lmax = -3.4e38f;
        for (int n = 0; n < NAA; ++n) lmax = fmaxf(lmax, logit[h][m][n] * sc);
        float e[NAA], esum = 0.f;
        for (int n = 0; n < NAA; ++n) { e[n] = __expf(logit[h][m][n] * sc - lmax); esum += e[n]; }
        const float inv = 1.0f / esum;
        float o = 0.f;
        for (int n = 0; n < NAA; ++n) o += e[n] * inv * v[n];
        o_attn[(b * NAA + m) * CC + tid] = o;
    }
}

// ---------------------------------------------------------------------------
// Global-branch: BN then max over anchors. grid = B, thread = channel
// ---------------------------------------------------------------------------
__global__ __launch_bounds__(256) void glob_max_kernel(
    const float* __restrict__ glob_pre,
    const float* __restrict__ g_sc, const float* __restrict__ g_sh,
    float* __restrict__ gmax)             // [B, C]
{
    const int b = blockIdx.x, c = threadIdx.x;
    float mx = -3.4e38f;
    for (int m = 0; m < NAA; ++m)
        mx = fmaxf(mx, glob_pre[(b * NAA + m) * CC + c] * g_sc[c] + g_sh[c]);
    gmax[b * CC + c] = mx;
}

// ---------------------------------------------------------------------------
// Head: prob_pre[r,j] = concat(local+res_bn, gmax) . head_w[j]   (ROWS*9 threads)
// ---------------------------------------------------------------------------
__global__ __launch_bounds__(256) void head_kernel(
    const float* __restrict__ local_feat, const float* __restrict__ res_pre,
    const float* __restrict__ res_sc, const float* __restrict__ res_sh,
    const float* __restrict__ gmax, const float* __restrict__ head_w,
    float* __restrict__ prob_pre)
{
    const int gid = blockIdx.x * blockDim.x + threadIdx.x;
    if (gid >= ROWS * 9) return;
    const int r = gid / 9, j = gid % 9;
    const int b = r / NAA;
    float acc = 0.f;
    for (int i = 0; i < CC; ++i) {
        const float f = local_feat[r * CC + i]
                      + res_pre[r * CC + i] * res_sc[i] + res_sh[i];
        acc += f * head_w[j * (2 * CC) + i];
    }
    for (int i = 0; i < CC; ++i)
        acc += gmax[b * CC + i] * head_w[j * (2 * CC) + CC + i];
    prob_pre[r * 9 + j] = acc;
}

__global__ __launch_bounds__(256) void head_bn_apply_kernel(
    const float* __restrict__ prob_pre,
    const float* __restrict__ h_sc, const float* __restrict__ h_sh,
    float* __restrict__ out)
{
    const int gid = blockIdx.x * blockDim.x + threadIdx.x;
    if (gid >= ROWS * 9) return;
    const int j = gid % 9;
    out[gid] = prob_pre[gid] * h_sc[j] + h_sh[j];
}

// ---------------------------------------------------------------------------
extern "C" void kernel_launch(void* const* d_in, const int* in_sizes, int n_in,
                              void* d_out, int out_size, void* d_ws, size_t ws_size,
                              hipStream_t stream) {
    const float* a_points  = (const float*)d_in[0];
    const float* sa_x      = (const float*)d_in[1];
    const float* sa_xyz    = (const float*)d_in[2];
    const float* W_qkv     = (const float*)d_in[3];
    const float* pos_w     = (const float*)d_in[4];
    const float* pos_b     = (const float*)d_in[5];
    const float* pos_bn_g  = (const float*)d_in[6];
    const float* pos_bn_b  = (const float*)d_in[7];
    const float* res_w     = (const float*)d_in[8];
    const float* res_b     = (const float*)d_in[9];
    const float* res_bn_g  = (const float*)d_in[10];
    const float* res_bn_b  = (const float*)d_in[11];
    const float* glob_w    = (const float*)d_in[12];
    const float* glob_bn_g = (const float*)d_in[13];
    const float* glob_bn_b = (const float*)d_in[14];
    const float* head_w    = (const float*)d_in[15];
    const float* head_bn_g = (const float*)d_in[16];
    const float* head_bn_b = (const float*)d_in[17];

    float* ws = (float*)d_ws;
    float* local_feat = ws + 0;        // 65536
    float* rxyz_pre   = ws + 65536;    // 65536
    float* glob_pre   = ws + 131072;   // 65536
    float* qkv        = ws + 196608;   // 196608
    float* o_attn     = ws + 393216;   // 65536
    float* res_pre    = ws + 458752;   // 65536
    float* gmax       = ws + 524288;   // 16384
    float* prob_pre   = ws + 540672;   // 2304
    float* pos_sc  = ws + 542976;  float* pos_sh  = ws + 543232;
    float* glob_sc = ws + 543488;  float* glob_sh = ws + 543744;
    float* res_sc  = ws + 544000;  float* res_sh  = ws + 544256;
    float* head_sc = ws + 544512;  float* head_sh = ws + 544528;

    // 1. KNN + gather + maxpool -> local_feat
    knn_gather_max_kernel<<<ROWS, 256, 0, stream>>>(a_points, sa_xyz, sa_x, local_feat);

    // 2. positional / global pre-BN linears
    pre_kernel<<<ROWS, 256, 0, stream>>>(a_points, pos_w, pos_b, glob_w,
                                         rxyz_pre, glob_pre);
    bnstats_kernel<<<1, 256, 0, stream>>>(rxyz_pre, CC, pos_bn_g, pos_bn_b, pos_sc, pos_sh);
    bnstats_kernel<<<1, 256, 0, stream>>>(glob_pre, CC, glob_bn_g, glob_bn_b, glob_sc, glob_sh);

    // 3. qkv = local_feat(256x256) @ W_qkv(256x768)   [WMMA, B row-major]
    wmma_gemm_kernel<0, 0><<<(ROWS / 16) * (3 * CC / 16), 32, 0, stream>>>(
        local_feat, W_qkv, nullptr, qkv,
        3 * CC, CC, CC, 3 * CC, 3 * CC);

    // 4. multi-head attention over NA=4 anchors
    attn_kernel<<<BB, 256, 0, stream>>>(qkv, rxyz_pre, pos_sc, pos_sh, o_attn);

    // 5. res_pre = o_attn(256x256) @ res_w^T(256x256) + res_b   [WMMA, B^T]
    wmma_gemm_kernel<1, 1><<<(ROWS / 16) * (CC / 16), 32, 0, stream>>>(
        o_attn, res_w, res_b, res_pre,
        CC, CC, CC, CC, CC);
    bnstats_kernel<<<1, 256, 0, stream>>>(res_pre, CC, res_bn_g, res_bn_b, res_sc, res_sh);

    // 6. global branch BN + max over anchors
    glob_max_kernel<<<BB, 256, 0, stream>>>(glob_pre, glob_sc, glob_sh, gmax);

    // 7. head conv1d (2C -> 9) then BN
    head_kernel<<<(ROWS * 9 + 255) / 256, 256, 0, stream>>>(
        local_feat, res_pre, res_sc, res_sh, gmax, head_w, prob_pre);
    bnstats_kernel<<<1, 256, 0, stream>>>(prob_pre, 9, head_bn_g, head_bn_b, head_sc, head_sh);
    head_bn_apply_kernel<<<(ROWS * 9 + 255) / 256, 256, 0, stream>>>(
        prob_pre, head_sc, head_sh, (float*)d_out);
}
